// MultiHeadAttention_73615739453791
// MI455X (gfx1250) — matrix operand order repaired and checked by reading, not compiled, not run on hardware
//
#include <hip/hip_runtime.h>
#include <hip/hip_bf16.h>

typedef _Float16 h8_t  __attribute__((ext_vector_type(8)));
typedef _Float16 h16_t __attribute__((ext_vector_type(16)));
typedef float    f8_t  __attribute__((ext_vector_type(8)));
typedef int      i4_t  __attribute__((ext_vector_type(4)));

#define EMB   1024
#define HEADS 16
#define HDIM  64
#define BATCH 2
#define SEQ   2048
#define MROWS (BATCH*SEQ)   // 4096 token rows

#define GLOBAL_AS __attribute__((address_space(1)))
#define LDS_AS    __attribute__((address_space(3)))

#if __has_builtin(__builtin_amdgcn_global_load_async_to_lds_b128)
#define HAVE_ASYNC_LDS 1
#else
#define HAVE_ASYNC_LDS 0
#endif

// ---------------------------------------------------------------------------
// Direct global->LDS copy on the CDNA5 async path (ASYNCcnt, no VGPR bounce).
// Builtin signature (from hipcc diagnostic): pointee type is a 16B int vector.
// Falls back to a VGPR round-trip copy if the builtin is unavailable.
// ---------------------------------------------------------------------------
__device__ __forceinline__ void async_copy16(_Float16* lds_dst, const _Float16* gsrc) {
#if HAVE_ASYNC_LDS
  __builtin_amdgcn_global_load_async_to_lds_b128(
      (GLOBAL_AS i4_t*)(uintptr_t)gsrc,
      (LDS_AS i4_t*)(unsigned)(uintptr_t)lds_dst,  // LDS offset = low 32 bits
      0, 0);
#else
  *(uint4*)lds_dst = *(const uint4*)gsrc;
#endif
}

__device__ __forceinline__ void wait_async0() {
#if HAVE_ASYNC_LDS
#if __has_builtin(__builtin_amdgcn_s_wait_asynccnt)
  __builtin_amdgcn_s_wait_asynccnt(0);
#else
  asm volatile("s_wait_asynccnt 0x0" ::: "memory");
#endif
#endif
}

// Same-wave LDS visibility (cross-lane) without a full workgroup barrier.
__device__ __forceinline__ void wait_ds0() {
  asm volatile("s_wait_dscnt 0x0" ::: "memory");
}

// ---------------------------------------------------------------------------
// f32 -> f16 downconvert (one-time; all heavy math runs on v_wmma f16 pipe)
// ---------------------------------------------------------------------------
__global__ __launch_bounds__(256) void cvt_f32_f16(const float* __restrict__ src,
                                                   _Float16* __restrict__ dst, int n) {
  int i = blockIdx.x * blockDim.x + threadIdx.x;
  if (i < n) dst[i] = (_Float16)src[i];
}

// Build a v16h A/B operand from two 16-byte LDS/global chunks.
__device__ __forceinline__ h16_t ld_a16(const _Float16* p0, const _Float16* p1) {
  h8_t lo = *(const h8_t*)p0;
  h8_t hi = *(const h8_t*)p1;
  return __builtin_shufflevector(lo, hi, 0,1,2,3,4,5,6,7,8,9,10,11,12,13,14,15);
}

// ---------------------------------------------------------------------------
// GEMM: C[M,N] = A[M,K] @ W[N,K]^T (+bias).  M=4096, N=K=1024.
// WG tile 128(M) x 64(N), BK=64, double-buffered LDS filled by async copies;
// 8 waves, each wave owns a 16-row strip and 4 N-subtiles -> 4 v8f accums.
// mode 0: f32 out [M,N]; mode 1: f16 -> [B*H,S,D]; mode 2: f16 -> [B*H,D,S].
// ---------------------------------------------------------------------------
#define BM 128
#define BN 64
#define BK 64
#define LDA 72
#define LDB 72

__global__ __launch_bounds__(256) void gemm_wmma(
    const _Float16* __restrict__ A, const _Float16* __restrict__ W,
    const float* __restrict__ bias,
    _Float16* __restrict__ dstH, float* __restrict__ dstF, int mode)
{
  __shared__ _Float16 Alds[2][BM * LDA];
  __shared__ _Float16 Blds[2][BN * LDB];
  const int K = EMB, N = EMB;
  const int n0 = blockIdx.x * BN;
  const int m0 = blockIdx.y * BM;
  const int tid = threadIdx.x;
  const int lane = tid & 31, wave = tid >> 5;
  const int l = lane & 15, grp = lane >> 4;

  auto load_tiles = [&](int kc, int buf) {
    #pragma unroll
    for (int u = 0; u < 4; ++u) {              // A tile: 128x64 halves, 1024 16B units
      int unit = tid + u * 256;
      int row = unit >> 3, seg = unit & 7;
      async_copy16(&Alds[buf][row * LDA + seg * 8],
                   &A[(size_t)(m0 + row) * K + kc + seg * 8]);
    }
    #pragma unroll
    for (int u = 0; u < 2; ++u) {              // B tile: 64x64 halves, 512 units
      int unit = tid + u * 256;
      int row = unit >> 3, seg = unit & 7;
      async_copy16(&Blds[buf][row * LDB + seg * 8],
                   &W[(size_t)(n0 + row) * K + kc + seg * 8]);
    }
  };

  f8_t acc[4] = {};

  load_tiles(0, 0);
  wait_async0();
  __syncthreads();

  int cur = 0;
  for (int kc = 0; kc < K; kc += BK) {
    if (kc + BK < K) load_tiles(kc + BK, cur ^ 1);   // prefetch next tile (async)

    #pragma unroll
    for (int k0 = 0; k0 < BK; k0 += 32) {
      // A operand (16x32): lane=M row; grp0 K 0..15, grp1 K 16..31 (contiguous)
      const _Float16* ap = &Alds[cur][(wave * 16 + l) * LDA + k0 + grp * 16];
      h16_t a = ld_a16(ap, ap + 8);
      #pragma unroll
      for (int t = 0; t < 4; ++t) {
        // B operand (32x16): lane=N col; grp0 K 0..15, grp1 K 16..31 (contiguous)
        const _Float16* bp = &Blds[cur][(t * 16 + l) * LDB + k0 + grp * 16];
        h16_t b = ld_a16(bp, bp + 8);
        acc[t] = __builtin_amdgcn_wmma_f32_16x16x32_f16(
            false, a, false, b, (short)0, acc[t], false, false);
      }
    }

    wait_async0();        // my prefetch landed
    __syncthreads();      // everyone done reading cur + filling nxt
    cur ^= 1;
  }

  // Epilogue. C layout: lane=col, VGPR r = row (grp0: r, grp1: 8+r).
  #pragma unroll
  for (int t = 0; t < 4; ++t) {
    int col = n0 + t * 16 + l;
    float bval = bias[col];
    #pragma unroll
    for (int r = 0; r < 8; ++r) {
      int row = m0 + wave * 16 + grp * 8 + r;
      float v = acc[t][r] + bval;
      if (mode == 0) {
        dstF[(size_t)row * N + col] = v;
      } else {
        int bb = row >> 11, s = row & (SEQ - 1);
        int hh = col >> 6,  d = col & (HDIM - 1);
        if (mode == 1)
          dstH[(((size_t)(bb * HEADS + hh)) * SEQ + s) * HDIM + d] = (_Float16)v;
        else
          dstH[(((size_t)(bb * HEADS + hh)) * HDIM + d) * SEQ + s] = (_Float16)v;
      }
    }
  }
}

// ---------------------------------------------------------------------------
// Fused flash attention.  Grid: (S/128, B*H).  8 waves; wave w owns a 16-row
// query strip.  Key blocks of 64, online softmax; K and pre-transposed V
// double-buffered in LDS via async copies; P (per-wave) does an LDS
// round-trip (C-layout -> A-layout) guarded by s_wait_dscnt only.
// ---------------------------------------------------------------------------
#define QB 128
#define KB 64
#define LDK 72
#define LDP 72

__global__ __launch_bounds__(256) void attn_wmma(
    const _Float16* __restrict__ Q,   // [BH,S,D]
    const _Float16* __restrict__ Km,  // [BH,S,D]
    const _Float16* __restrict__ Vt,  // [BH,D,S]
    _Float16* __restrict__ Attn)      // [B*S, E] f16
{
  __shared__ _Float16 Klds[2][KB * LDK];
  __shared__ _Float16 Vlds[2][HDIM * LDK];
  __shared__ _Float16 Plds[8 * 16 * LDP];

  const int bh = blockIdx.y;
  const int qb = blockIdx.x * QB;
  const int tid = threadIdx.x, lane = tid & 31, wave = tid >> 5;
  const int l = lane & 15, grp = lane >> 4;
  const float scale = 1.0f / 32.0f;  // 1/sqrt(EMB) per reference

  auto load_kv = [&](int k0b, int buf) {
    #pragma unroll
    for (int u = 0; u < 2; ++u) {
      int unit = tid + u * 256;
      int row = unit >> 3, seg = unit & 7;
      async_copy16(&Klds[buf][row * LDK + seg * 8],
                   &Km[((size_t)bh * SEQ + k0b + row) * HDIM + seg * 8]);
      async_copy16(&Vlds[buf][row * LDK + seg * 8],
                   &Vt[((size_t)bh * HDIM + row) * SEQ + k0b + seg * 8]);
    }
  };

  // Q strip -> A-operand registers (d 0..31 and 32..63), held for whole kernel
  const int qrow = qb + wave * 16 + l;
  const _Float16* qp = Q + ((size_t)bh * SEQ + qrow) * HDIM + grp * 16;
  h16_t aQ0 = ld_a16(qp,      qp + 8);
  h16_t aQ1 = ld_a16(qp + 32, qp + 40);

  float mrow[8], lrow[8];
  f8_t o[4] = {};
  #pragma unroll
  for (int r = 0; r < 8; ++r) { mrow[r] = -1e30f; lrow[r] = 0.0f; }

  _Float16* Pw = &Plds[wave * 16 * LDP];

  load_kv(0, 0);
  wait_async0();
  __syncthreads();

  int cur = 0;
  for (int k0b = 0; k0b < SEQ; k0b += KB) {
    if (k0b + KB < SEQ) load_kv(k0b + KB, cur ^ 1);   // async prefetch next block

    // energy strip 16x64: E = Q @ K^T ; B operand B[d][key] = K[key][d]
    f8_t e[4];
    #pragma unroll
    for (int t = 0; t < 4; ++t) {
      f8_t et = {};
      const _Float16* bp = &Klds[cur][(t * 16 + l) * LDK + grp * 16];
      h16_t b0 = ld_a16(bp,      bp + 8);       // d 0..31
      et = __builtin_amdgcn_wmma_f32_16x16x32_f16(false, aQ0, false, b0, (short)0, et, false, false);
      h16_t b1 = ld_a16(bp + 32, bp + 40);      // d 32..63
      et = __builtin_amdgcn_wmma_f32_16x16x32_f16(false, aQ1, false, b1, (short)0, et, false, false);
      e[t] = et;
    }

    // online softmax per q-row (16-lane group holds 16 key columns of one row)
    #pragma unroll
    for (int r = 0; r < 8; ++r) {
      float e0 = e[0][r] * scale, e1 = e[1][r] * scale;
      float e2 = e[2][r] * scale, e3 = e[3][r] * scale;
      float bm = fmaxf(fmaxf(e0, e1), fmaxf(e2, e3));
      #pragma unroll
      for (int off = 8; off >= 1; off >>= 1) bm = fmaxf(bm, __shfl_xor(bm, off, 32));
      float mn = fmaxf(mrow[r], bm);
      float cf = __expf(mrow[r] - mn);
      mrow[r] = mn;
      float p0 = __expf(e0 - mn), p1 = __expf(e1 - mn);
      float p2 = __expf(e2 - mn), p3 = __expf(e3 - mn);
      float rs = p0 + p1 + p2 + p3;
      #pragma unroll
      for (int off = 8; off >= 1; off >>= 1) rs += __shfl_xor(rs, off, 32);
      lrow[r] = lrow[r] * cf + rs;
      o[0][r] *= cf; o[1][r] *= cf; o[2][r] *= cf; o[3][r] *= cf;
      e[0][r] = p0; e[1][r] = p1; e[2][r] = p2; e[3][r] = p3;
    }

    // P: C-layout regs -> per-wave LDS [16 rows][64 keys]
    #pragma unroll
    for (int t = 0; t < 4; ++t)
      #pragma unroll
      for (int r = 0; r < 8; ++r)
        Pw[(grp * 8 + r) * LDP + t * 16 + l] = (_Float16)e[t][r];
    wait_ds0();   // same-wave cross-lane LDS visibility; no WG barrier needed

    // P back as A operand (keys 0..31 / 32..63)
    const _Float16* pp = &Pw[l * LDP + grp * 16];
    h16_t aP0 = ld_a16(pp,      pp + 8);
    h16_t aP1 = ld_a16(pp + 32, pp + 40);

    // o += P @ V ; B operand B[key][d] = Vt[d][key] -> contiguous LDS rows
    #pragma unroll
    for (int t = 0; t < 4; ++t) {
      const _Float16* vp = &Vlds[cur][(t * 16 + l) * LDK + grp * 16];
      h16_t bv0 = ld_a16(vp,      vp + 8);      // keys 0..31
      o[t] = __builtin_amdgcn_wmma_f32_16x16x32_f16(false, aP0, false, bv0, (short)0, o[t], false, false);
      h16_t bv1 = ld_a16(vp + 32, vp + 40);     // keys 32..63
      o[t] = __builtin_amdgcn_wmma_f32_16x16x32_f16(false, aP1, false, bv1, (short)0, o[t], false, false);
    }

    wait_async0();        // my K/V prefetch landed
    __syncthreads();      // all waves done with buffer `cur`
    cur ^= 1;
  }

  // normalize and scatter to [B*S, E] f16 for the output projection
  const int bb = bh >> 4, hh = bh & (HEADS - 1);
  #pragma unroll
  for (int t = 0; t < 4; ++t) {
    int ecol = hh * HDIM + t * 16 + l;
    #pragma unroll
    for (int r = 0; r < 8; ++r) {
      int s = qb + wave * 16 + grp * 8 + r;
      float v = o[t][r] / lrow[r];
      Attn[((size_t)(bb * SEQ + s)) * EMB + ecol] = (_Float16)v;
    }
  }
}

// ---------------------------------------------------------------------------
extern "C" void kernel_launch(void* const* d_in, const int* in_sizes, int n_in,
                              void* d_out, int out_size, void* d_ws, size_t ws_size,
                              hipStream_t stream) {
  const float* x  = (const float*)d_in[0];
  const float* Wq = (const float*)d_in[1];
  const float* bq = (const float*)d_in[2];
  const float* Wk = (const float*)d_in[3];
  const float* bk = (const float*)d_in[4];
  const float* Wv = (const float*)d_in[5];
  const float* bv = (const float*)d_in[6];
  const float* Wo = (const float*)d_in[7];
  const float* bo = (const float*)d_in[8];

  char* w = (char*)d_ws;
  _Float16* Xh  = (_Float16*)w; w += (size_t)MROWS * EMB * 2;            // 8 MB
  _Float16* Wqh = (_Float16*)w; w += (size_t)EMB * EMB * 2;              // 2 MB
  _Float16* Wkh = (_Float16*)w; w += (size_t)EMB * EMB * 2;
  _Float16* Wvh = (_Float16*)w; w += (size_t)EMB * EMB * 2;
  _Float16* Woh = (_Float16*)w; w += (size_t)EMB * EMB * 2;
  _Float16* Qh  = (_Float16*)w; w += (size_t)MROWS * EMB * 2;            // [BH,S,D]
  _Float16* Kh  = (_Float16*)w; w += (size_t)MROWS * EMB * 2;            // [BH,S,D]
  _Float16* Vth = (_Float16*)w; w += (size_t)MROWS * EMB * 2;            // [BH,D,S]
  _Float16* Ah  = (_Float16*)w; w += (size_t)MROWS * EMB * 2;            // [B*S,E]

  const int nX = MROWS * EMB, nW = EMB * EMB;
  cvt_f32_f16<<<nX / 256, 256, 0, stream>>>(x,  Xh,  nX);
  cvt_f32_f16<<<nW / 256, 256, 0, stream>>>(Wq, Wqh, nW);
  cvt_f32_f16<<<nW / 256, 256, 0, stream>>>(Wk, Wkh, nW);
  cvt_f32_f16<<<nW / 256, 256, 0, stream>>>(Wv, Wvh, nW);
  cvt_f32_f16<<<nW / 256, 256, 0, stream>>>(Wo, Woh, nW);

  dim3 gg(EMB / BN, MROWS / BM);     // (16, 32)
  gemm_wmma<<<gg, 256, 0, stream>>>(Xh, Wqh, bq, Qh,  nullptr, 1);
  gemm_wmma<<<gg, 256, 0, stream>>>(Xh, Wkh, bk, Kh,  nullptr, 1);
  gemm_wmma<<<gg, 256, 0, stream>>>(Xh, Wvh, bv, Vth, nullptr, 2);

  dim3 ga(SEQ / QB, BATCH * HEADS);  // (16, 32)
  attn_wmma<<<ga, 256, 0, stream>>>(Qh, Kh, Vth, Ah);

  gemm_wmma<<<gg, 256, 0, stream>>>(Ah, Woh, bo, nullptr, (float*)d_out, 0);
}
